// WindowAttention_ACAM_12421045420294
// MI455X (gfx1250) — matrix operand order repaired and checked by reading, not compiled, not run on hardware
//
#include <hip/hip_runtime.h>
#include <math.h>

// ---------------------------------------------------------------------------
// Shapes: B=2, H=W=64, N=4096, C=96, C12=8, SCALE = 8^-0.5
// ---------------------------------------------------------------------------
#define SCALE_F 0.35355339059327373f

typedef __attribute__((ext_vector_type(16))) _Float16 v16h;
typedef __attribute__((ext_vector_type(8)))  _Float16 v8h;
typedef __attribute__((ext_vector_type(8)))  float    v8f;

__device__ __forceinline__ v8f wmma_f16(v16h a, v16h b, v8f c) {
  // D = A(16x32,f16) * B(32x16,f16) + C(16x16,f32)
  return __builtin_amdgcn_wmma_f32_16x16x32_f16(false, a, false, b, (short)0, c,
                                                false, false);
}

// Wave-local LDS RAW fence: DS ops are in-order per wave; wait for stores to
// land before re-reading the per-wave staging tile. (CDNA5 split counters.)
__device__ __forceinline__ void wave_lds_fence() {
  asm volatile("s_wait_dscnt 0x0" ::: "memory");
}

// ---------------------------------------------------------------------------
// DPP16 all-reduce across a 16-lane group (no LDS, no counter waits).
// Steps: xor1 (quad_perm 1,0,3,2), xor2 (quad_perm 2,3,0,1),
//        row_half_mirror (merges 4-groups), row_mirror (merges 8-groups).
// ---------------------------------------------------------------------------
template<int CTRL>
__device__ __forceinline__ float dpp_mov_f32(float x) {
  return __int_as_float(__builtin_amdgcn_update_dpp(
      0, __float_as_int(x), CTRL, 0xf, 0xf, true));
}

__device__ __forceinline__ float row16_max(float x) {
  x = fmaxf(x, dpp_mov_f32<0xB1>(x));    // quad_perm [1,0,3,2]
  x = fmaxf(x, dpp_mov_f32<0x4E>(x));    // quad_perm [2,3,0,1]
  x = fmaxf(x, dpp_mov_f32<0x141>(x));   // row_half_mirror
  x = fmaxf(x, dpp_mov_f32<0x140>(x));   // row_mirror
  return x;
}

__device__ __forceinline__ float row16_sum(float x) {
  x += dpp_mov_f32<0xB1>(x);
  x += dpp_mov_f32<0x4E>(x);
  x += dpp_mov_f32<0x141>(x);
  x += dpp_mov_f32<0x140>(x);
  return x;
}

// ---------------------------------------------------------------------------
// Kernel 1: 1x1 convs -> q/k/v for CAM / PAM / CHAM (9 GEMVs of (8x96)@(96x4096))
// grid = (128, matrix 0..8, batch 0..1), 256 thr. PAM outputs stored transposed
// as (4096 x 8) row-major.
// ---------------------------------------------------------------------------
struct QKVParams {
  const float* x;
  const float* w[9];
  const float* bias[9];
  float* out[9];
};

__global__ __launch_bounds__(256) void qkv_conv1x1(QKVParams P) {
  const int m = blockIdx.y;       // 0..8 : cam q,k,v / pam q,k,v / cham q,k,v
  const int b = blockIdx.z;
  const int idx = blockIdx.x * 256 + threadIdx.x;   // 0..32767
  const int o = idx >> 12;        // 0..7 (uniform per block: 16 blocks per o)
  const int p = idx & 4095;       // spatial
  const float* w = P.w[m];
  const float* x = P.x + (long)b * 393216;
  float acc = P.bias[m][o];
  #pragma unroll 8
  for (int c = 0; c < 96; ++c)
    acc = fmaf(w[o * 96 + c], x[c * 4096 + p], acc);
  float* out = P.out[m] + (long)b * 32768;
  if (m >= 3 && m < 6) out[p * 8 + o] = acc;    // PAM: transposed (N x 8)
  else                 out[o * 4096 + p] = acc; // (8 x N) == (512 x 64)
}

// ---------------------------------------------------------------------------
// Kernel 2: CAM channel attention (8x8 attn over N=4096). Tiny -> scalar VALU.
// ---------------------------------------------------------------------------
__global__ __launch_bounds__(256) void cam_attn(const float* __restrict__ q,
                                                const float* __restrict__ k,
                                                const float* __restrict__ v,
                                                float* __restrict__ y) {
  const int b = blockIdx.x;
  const float* Q = q + (long)b * 32768;
  const float* K = k + (long)b * 32768;
  const float* V = v + (long)b * 32768;
  float* Y = y + (long)b * 32768;
  __shared__ float part[256];
  __shared__ float A[64];
  const int tid = threadIdx.x;
  const int pair = tid >> 2, seg = tid & 3;   // 64 (c,d) pairs x 4 segments
  const int c = pair >> 3, d = pair & 7;
  float s = 0.f;
  const int n0 = seg * 1024;
  for (int n = n0; n < n0 + 1024; ++n) s = fmaf(Q[c * 4096 + n], K[d * 4096 + n], s);
  part[tid] = s;
  __syncthreads();
  if (tid < 64)
    A[tid] = (part[tid * 4] + part[tid * 4 + 1] + part[tid * 4 + 2] + part[tid * 4 + 3]) * SCALE_F;
  __syncthreads();
  if (tid < 8) {       // softmax row `tid`
    float mx = -3.0e38f;
    #pragma unroll
    for (int j = 0; j < 8; ++j) mx = fmaxf(mx, A[tid * 8 + j]);
    float sm = 0.f;
    #pragma unroll
    for (int j = 0; j < 8; ++j) { float e = __expf(A[tid * 8 + j] - mx); A[tid * 8 + j] = e; sm += e; }
    float inv = 1.f / sm;
    #pragma unroll
    for (int j = 0; j < 8; ++j) A[tid * 8 + j] *= inv;
  }
  __syncthreads();
  for (int i = tid; i < 32768; i += 256) {
    int cc = i >> 12, n = i & 4095;
    float acc = 0.f;
    #pragma unroll
    for (int j = 0; j < 8; ++j) acc = fmaf(A[cc * 8 + j], V[j * 4096 + n], acc);
    Y[i] = acc;
  }
}

// ---------------------------------------------------------------------------
// Kernel 3: generic flash attention with WMMA f16->f32.
//   Q (Nr x D), K (Nc x D), V (Nc x D) row-major f32; Out (Nr x D) f32.
//   One wave per 16-row Q tile, 8 waves/block (128 rows). 32-key blocks are
//   staged into LDS as f16 (K row-major, V transposed) with zero padding, so
//   every WMMA fragment load is an unconditional, contiguous 32B LDS read.
//   Online softmax lives in registers; row reductions are DPP16 butterflies
//   (a C-layout row's 16 columns occupy exactly one 16-lane group). P goes
//   through a per-wave LDS tile (wave-local dscnt fence, no block barrier)
//   to convert C-layout -> A-fragment layout.
// ---------------------------------------------------------------------------
template<int D>
__global__ __launch_bounds__(256) void flash_attn(
    const float* __restrict__ Qb, const float* __restrict__ Kb,
    const float* __restrict__ Vb, float* __restrict__ Ob,
    int Nc, long qBatch, long kBatch, long vBatch, long oBatch, float scale) {
  constexpr int KCH  = (D + 31) / 32;        // 32-wide K chunks for score GEMM
  constexpr int KPAD = KCH * 32;             // padded K extent (32 or 64)
  constexpr int DP   = (D < 16) ? 16 : D;    // padded value-dim extent
  constexpr int NT   = DP / 16;              // 16-col output tiles

  __shared__ alignas(32) _Float16 sK16[32][KPAD];  // keys: [key][k], f16
  __shared__ alignas(32) _Float16 sVt[DP][32];     // values transposed: [d][key]
  __shared__ alignas(32) _Float16 sP[8][16][32];   // per-wave P staging

  const int lane = threadIdx.x & 31;
  const int wv   = threadIdx.x >> 5;
  const int b    = blockIdx.y;
  const float* Q = Qb + (long)b * qBatch;
  const float* K = Kb + (long)b * kBatch;
  const float* V = Vb + (long)b * vBatch;
  float* O       = Ob + (long)b * oBatch;
  const int r0     = blockIdx.x * 128 + wv * 16;
  const int lane15 = lane & 15;
  const int halfHi = (lane & 16) ? 1 : 0;

  // zero-fill LDS pads once (covers D=8 padding; cheap, pre-loop)
  for (int i = threadIdx.x; i < 32 * KPAD; i += 256) (&sK16[0][0])[i] = (_Float16)0.f;
  for (int i = threadIdx.x; i < DP * 32; i += 256) (&sVt[0][0])[i] = (_Float16)0.f;

  // Q A-fragments (contiguous k = 16*half + e, same order as B), softmax scale folded in.
  v16h aq[KCH];
  #pragma unroll
  for (int c = 0; c < KCH; ++c) {
    #pragma unroll
    for (int e = 0; e < 16; ++e) {
      int kk = halfHi * 16 + e;
      int kidx = c * 32 + kk;
      float qv = (kidx < D) ? Q[(long)(r0 + lane15) * D + kidx] * scale : 0.f;
      aq[c][e] = (_Float16)qv;
    }
  }

  float mrun[8], lrun[8];
  v8f acc[NT];
  #pragma unroll
  for (int j = 0; j < 8; ++j) { mrun[j] = -3.0e38f; lrun[j] = 0.f; }
  #pragma unroll
  for (int t = 0; t < NT; ++t)
    #pragma unroll
    for (int j = 0; j < 8; ++j) acc[t][j] = 0.f;

  for (int m0 = 0; m0 < Nc; m0 += 32) {
    __syncthreads();
    // cooperative f16 staging of 32 keys + transposed values
    for (int i = threadIdx.x; i < 32 * D; i += 256) {
      int r = i / D, cc = i % D;
      sK16[r][cc] = (_Float16)K[(long)(m0 + r) * D + cc];
      sVt[cc][r]  = (_Float16)V[(long)(m0 + r) * D + cc];
    }
    __syncthreads();

    // scores: S(16x32) = Q_tile(16xD) * K_blk^T, two 16x16 accumulators
    v8f s0, s1;
    #pragma unroll
    for (int j = 0; j < 8; ++j) { s0[j] = 0.f; s1[j] = 0.f; }
    #pragma unroll
    for (int c = 0; c < KCH; ++c) {
      // B layout: lane half selects K 0-15 / 16-31; column = lane15
      v16h b0 = *(const v16h*)(&sK16[lane15][c * 32 + halfHi * 16]);
      v16h b1 = *(const v16h*)(&sK16[16 + lane15][c * 32 + halfHi * 16]);
      s0 = wmma_f16(aq[c], b0, s0);
      s1 = wmma_f16(aq[c], b1, s1);
    }

    // online softmax (row j of this wave's tile; columns = one 16-lane group)
    float corr[8];
    #pragma unroll
    for (int j = 0; j < 8; ++j) {
      float a0 = s0[j], a1 = s1[j];
      float mb = row16_max(fmaxf(a0, a1));
      float mn = fmaxf(mrun[j], mb);
      float cr = __expf(mrun[j] - mn);
      float p0 = __expf(a0 - mn);
      float p1 = __expf(a1 - mn);
      float sb = row16_sum(p0 + p1);
      lrun[j] = lrun[j] * cr + sb;
      mrun[j] = mn;
      corr[j] = cr;
      int row = j + halfHi * 8;                    // C-layout row id
      sP[wv][row][lane15]      = (_Float16)p0;
      sP[wv][row][16 + lane15] = (_Float16)p1;
    }
    #pragma unroll
    for (int t = 0; t < NT; ++t)
      #pragma unroll
      for (int j = 0; j < 8; ++j) acc[t][j] *= corr[j];

    wave_lds_fence();   // per-wave tile: DS in-order, just wait for stores

    // P as A-fragment: two contiguous 16B chunks per lane
    v8h plo = *(const v8h*)(&sP[wv][lane15][halfHi * 16]);
    v8h phi = *(const v8h*)(&sP[wv][lane15][halfHi * 16 + 8]);
    v16h pa = __builtin_shufflevector(plo, phi, 0, 1, 2, 3, 4, 5, 6, 7,
                                      8, 9, 10, 11, 12, 13, 14, 15);
    // O += P(16x32) * V_blk(32xD); V staged transposed -> contiguous B frags
    #pragma unroll
    for (int t = 0; t < NT; ++t) {
      v16h vb = *(const v16h*)(&sVt[t * 16 + lane15][halfHi * 16]);
      acc[t] = wmma_f16(pa, vb, acc[t]);
    }
  }

  // normalize + store (one reciprocal per row, not one divide per element)
  #pragma unroll
  for (int j = 0; j < 8; ++j) {
    float inv = 1.f / lrun[j];
    int row = r0 + j + halfHi * 8;
    #pragma unroll
    for (int t = 0; t < NT; ++t) {
      int col = t * 16 + lane15;
      if (col < D) O[(long)row * D + col] = acc[t][j] * inv;
    }
  }
}

// ---------------------------------------------------------------------------
// Kernel 4: grouped 3x3 conv -> exact GELU -> (1+g)*gamma scaled, all four
// branches fused and summed directly into out (B, N, C).
// ---------------------------------------------------------------------------
__device__ __forceinline__ float conv3x3_at(const float* __restrict__ ych,
                                            int h, int w, const float* __restrict__ wt) {
  float s = 0.f;
  #pragma unroll
  for (int dy = -1; dy <= 1; ++dy) {
    int hh = h + dy;
    if ((unsigned)hh >= 64u) continue;
    #pragma unroll
    for (int dx = -1; dx <= 1; ++dx) {
      int ww = w + dx;
      if ((unsigned)ww >= 64u) continue;
      s = fmaf(wt[(dy + 1) * 3 + (dx + 1)], ych[hh * 64 + ww], s);
    }
  }
  return s;
}

__device__ __forceinline__ float gelu_exact(float x) {
  return 0.5f * x * (1.f + erff(x * 0.70710678118654752f));
}

struct ProjParams {
  const float *y1, *y2, *y3, *y4;
  const float *w1, *w2, *w3, *w4;
  const float *cg1, *cg2, *cg3, *cg4;   // per-branch gamma g
  const float *gm1, *gm2, *gm3, *gm4;   // gamma1..4
  float* out;
};

__global__ __launch_bounds__(256) void proj_combine(ProjParams P) {
  const int gid = blockIdx.x * 256 + threadIdx.x;  // < 786432
  const int p = gid & 4095;
  const int rem = gid >> 12;
  const int o = rem % 96;
  const int b = rem / 96;
  const int h = p >> 6, w = p & 63;
  const int g8 = o / 12;    // group channel for G=8 branches
  float t1 = gelu_exact(conv3x3_at(P.y1 + (long)b * 32768 + g8 * 4096, h, w, P.w1 + o * 9));
  float t2 = gelu_exact(conv3x3_at(P.y2 + (long)b * 32768 + g8 * 4096, h, w, P.w2 + o * 9));
  float t3 = gelu_exact(conv3x3_at(P.y3 + (long)b * 32768 + g8 * 4096, h, w, P.w3 + o * 9));
  float t4 = gelu_exact(conv3x3_at(P.y4 + (long)b * 393216 + o * 4096, h, w, P.w4 + o * 9));
  float acc = P.gm1[0] * (1.f + P.cg1[0]) * t1
            + P.gm2[0] * (1.f + P.cg2[0]) * t2
            + P.gm3[0] * (1.f + P.cg3[0]) * t3
            + P.gm4[0] * (1.f + P.cg4[0]) * t4;
  P.out[(long)b * 393216 + (long)p * 96 + o] = acc;
}

// ---------------------------------------------------------------------------
extern "C" void kernel_launch(void* const* d_in, const int* in_sizes, int n_in,
                              void* d_out, int out_size, void* d_ws, size_t ws_size,
                              hipStream_t stream) {
  (void)in_sizes; (void)n_in; (void)out_size; (void)ws_size;
  auto in = [&](int i) { return (const float*)d_in[i]; };
  const float* x = in(0);

  float* ws = (float*)d_ws;
  size_t off = 0;
  auto carve = [&](size_t n) { float* pp = ws + off; off += n; return pp; };
  float* q1 = carve(65536); float* k1 = carve(65536); float* v1 = carve(65536);
  float* q2 = carve(65536); float* k2 = carve(65536); float* v2 = carve(65536);
  float* q3 = carve(65536); float* k3 = carve(65536); float* v3 = carve(65536);
  float* y1 = carve(65536); float* y2 = carve(65536); float* y3 = carve(65536);
  float* y4 = carve(786432);

  QKVParams qp;
  qp.x = x;
  qp.w[0] = in(1);  qp.bias[0] = in(2);
  qp.w[1] = in(3);  qp.bias[1] = in(4);
  qp.w[2] = in(5);  qp.bias[2] = in(6);
  qp.w[3] = in(9);  qp.bias[3] = in(10);
  qp.w[4] = in(11); qp.bias[4] = in(12);
  qp.w[5] = in(13); qp.bias[5] = in(14);
  qp.w[6] = in(17); qp.bias[6] = in(18);
  qp.w[7] = in(19); qp.bias[7] = in(20);
  qp.w[8] = in(21); qp.bias[8] = in(22);
  qp.out[0] = q1; qp.out[1] = k1; qp.out[2] = v1;
  qp.out[3] = q2; qp.out[4] = k2; qp.out[5] = v2;
  qp.out[6] = q3; qp.out[7] = k3; qp.out[8] = v3;
  qkv_conv1x1<<<dim3(128, 9, 2), 256, 0, stream>>>(qp);

  cam_attn<<<2, 256, 0, stream>>>(q1, k1, v1, y1);

  // PAM: seq 4096, d=8
  flash_attn<8><<<dim3(32, 2), 256, 0, stream>>>(q2, k2, v2, y2, 4096,
                                                 32768, 32768, 32768, 32768, SCALE_F);
  // CHAM: seq 512, d=64
  flash_attn<64><<<dim3(4, 2), 256, 0, stream>>>(q3, k3, v3, y3, 512,
                                                 32768, 32768, 32768, 32768, SCALE_F);
  // CWAM: self-attention on x viewed (6144 x 64)
  flash_attn<64><<<dim3(48, 2), 256, 0, stream>>>(x, x, x, y4, 6144,
                                                  393216, 393216, 393216, 393216, SCALE_F);

  ProjParams pp;
  pp.y1 = y1; pp.y2 = y2; pp.y3 = y3; pp.y4 = y4;
  pp.w1 = in(7);  pp.w2 = in(15); pp.w3 = in(23); pp.w4 = in(25);
  pp.cg1 = in(8); pp.cg2 = in(16); pp.cg3 = in(24); pp.cg4 = in(26);
  pp.gm1 = in(27); pp.gm2 = in(28); pp.gm3 = in(29); pp.gm4 = in(30);
  pp.out = (float*)d_out;
  proj_combine<<<3072, 256, 0, stream>>>(pp);
}
